// AttentionDilated_55628416418534
// MI455X (gfx1250) — matrix operand order repaired and checked by reading, not compiled, not run on hardware
//
#include <hip/hip_runtime.h>

typedef __bf16 bf16_t;
typedef __attribute__((ext_vector_type(8)))  __bf16 v8bf;
typedef __attribute__((ext_vector_type(16))) __bf16 v16bf;
typedef __attribute__((ext_vector_type(8)))  float  v8f;

// Problem constants (match reference)
constexpr int BB = 8, SS = 2048, FF = 512, UU = 512, PP = 256, DIL = 4;
constexpr float SCALE = 0.04419417382415922f; // 1/sqrt(512)

__device__ __forceinline__ int lane_id() { return (int)(threadIdx.x & 31u); }

// ---- WMMA fragment loaders (wave32, 16x16x32 bf16) -------------------------
// A: 16x32 tile from row-major [M][K] memory (ISA 7.12.2 16-bit A layout).
__device__ __forceinline__ v16bf load_frag_a(const bf16_t* __restrict__ base,
                                             int ld, int m0, int k0) {
  int l = lane_id();
  int m = m0 + (l & 15);
  int g = l >> 4;
  const bf16_t* p = base + (size_t)m * ld + k0;
  v8bf lo = *(const v8bf*)(p + g * 8);        // K = k0 + g*8 .. +7
  v8bf hi = *(const v8bf*)(p + 16 + g * 8);   // K = k0 + 16 + g*8 .. +7
  return __builtin_shufflevector(lo, hi, 0,1,2,3,4,5,6,7,8,9,10,11,12,13,14,15);
}

// B: 32x16 tile where memory is N-major: mem[n*ld + k] supplies B[k][n].
__device__ __forceinline__ v16bf load_frag_b_nmajor(const bf16_t* __restrict__ base,
                                                    int ld, int k0, int n0) {
  int l = lane_id();
  int n = n0 + (l & 15);
  int g = l >> 4;
  const bf16_t* p = base + (size_t)n * ld + k0;
  v8bf lo = *(const v8bf*)(p + g * 8);
  v8bf hi = *(const v8bf*)(p + 16 + g * 8);
  return __builtin_shufflevector(lo, hi, 0,1,2,3,4,5,6,7,8,9,10,11,12,13,14,15);
}

__device__ __forceinline__ v8f wmma_bf16(v16bf a, v16bf b, v8f c) {
  return __builtin_amdgcn_wmma_f32_16x16x32_bf16(false, a, false, b,
                                                 (short)0, c, false, false);
}

// ---- Register-blocked GEMM core: MT x NT tiles of 16x16 per wave -----------
template <int MT, int NT>
__device__ __forceinline__ void gemm_core(const bf16_t* __restrict__ A, int lda,
                                          const bf16_t* __restrict__ Bn, int ldb,
                                          int K, int m0, int n0, v8f c[MT][NT]) {
  int l = lane_id();
  for (int k0 = 0; k0 < K; k0 += 32) {
    // Unconditional near-scope prefetch of the A stream one cacheline ahead;
    // tail overrun lands in adjacent rows of the same ws buffer (harmless).
    __builtin_prefetch(A + (size_t)(m0 + (l & 15)) * lda + k0 + 128, 0, 3);
    v16bf a[MT];
    v16bf b[NT];
#pragma unroll
    for (int i = 0; i < MT; ++i) a[i] = load_frag_a(A, lda, m0 + 16 * i, k0);
#pragma unroll
    for (int j = 0; j < NT; ++j) b[j] = load_frag_b_nmajor(Bn, ldb, k0, n0 + 16 * j);
#pragma unroll
    for (int i = 0; i < MT; ++i)
#pragma unroll
      for (int j = 0; j < NT; ++j)
        c[i][j] = wmma_bf16(a[i], b[j], c[i][j]);
  }
}

// ---- Prep kernels ----------------------------------------------------------
__global__ void __launch_bounds__(256)
pack_x_kernel(const float* __restrict__ in, bf16_t* __restrict__ outp, size_t n) {
  size_t i = (size_t)blockIdx.x * blockDim.x + threadIdx.x;
  size_t stride = (size_t)gridDim.x * blockDim.x;
  for (; i < n; i += stride) outp[i] = (bf16_t)in[i];
}

// in: [R][C] f32 -> out: [C][R] bf16
__global__ void __launch_bounds__(256)
transpose_w_kernel(const float* __restrict__ in, bf16_t* __restrict__ outp,
                   int R, int C) {
  size_t n = (size_t)R * C;
  size_t i = (size_t)blockIdx.x * blockDim.x + threadIdx.x;
  size_t stride = (size_t)gridDim.x * blockDim.x;
  for (; i < n; i += stride) {
    int co = (int)(i / R), ro = (int)(i % R);
    outp[i] = (bf16_t)in[(size_t)ro * C + co];
  }
}

// x [B][S][F] f32 -> xT [B][F][S] bf16
__global__ void __launch_bounds__(256)
transpose_x_kernel(const float* __restrict__ x, bf16_t* __restrict__ xT) {
  size_t n = (size_t)BB * FF * SS;
  size_t i = (size_t)blockIdx.x * blockDim.x + threadIdx.x;
  size_t stride = (size_t)gridDim.x * blockDim.x;
  for (; i < n; i += stride) {
    int b = (int)(i / ((size_t)FF * SS));
    int r = (int)(i % ((size_t)FF * SS));
    int f = r / SS, t = r % SS;
    xT[i] = (bf16_t)x[((size_t)b * SS + t) * FF + f];
  }
}

// gathered query rows: xq[b][p][f] = x[b][positions[b][p]][f] (bf16)
__global__ void __launch_bounds__(256)
gather_xq_kernel(const float* __restrict__ x, const int* __restrict__ positions,
                 bf16_t* __restrict__ xq) {
  size_t n = (size_t)BB * PP * FF;
  size_t i = (size_t)blockIdx.x * blockDim.x + threadIdx.x;
  size_t stride = (size_t)gridDim.x * blockDim.x;
  for (; i < n; i += stride) {
    int b = (int)(i / ((size_t)PP * FF));
    int r = (int)(i % ((size_t)PP * FF));
    int p = r / FF, f = r % FF;
    int s = positions[b * PP + p];
    xq[i] = (bf16_t)x[((size_t)b * SS + s) * FF + f];
  }
}

// ---- GEMM kernels (one wave per 32x64 macro-tile) --------------------------
// out[M][512] (bf16) = A[M][512] @ Bn(N-major wT[n][k]) + bias[n]; N=K=512.
__global__ void __launch_bounds__(256)
gemm_bias_bf16(const bf16_t* __restrict__ A, const bf16_t* __restrict__ Bn,
               const float* __restrict__ bias, bf16_t* __restrict__ outp,
               int M) {
  constexpr int MT = 2, NT = 4;
  constexpr int N = UU, K = FF;
  int w = blockIdx.x * 8 + (int)(threadIdx.x >> 5);
  constexpr int Nt = N / (16 * NT);                 // 8
  if (w >= (M / (16 * MT)) * Nt) return;
  int m0 = (w / Nt) * 16 * MT, n0 = (w % Nt) * 16 * NT;
  v8f c[MT][NT] = {};
  gemm_core<MT, NT>(A, K, Bn, K, K, m0, n0, c);
  int l = lane_id();
  int g = l >> 4;
#pragma unroll
  for (int j = 0; j < NT; ++j) {
    int n = n0 + 16 * j + (l & 15);
    float bv = bias[n];
#pragma unroll
    for (int i = 0; i < MT; ++i)
#pragma unroll
      for (int r = 0; r < 8; ++r) {
        int m = m0 + 16 * i + r + 8 * g;
        outp[(size_t)m * N + n] = (bf16_t)(c[i][j][r] + bv);
      }
  }
}

// scores[b][p][t] = mask * (q_p . k_t) * SCALE  (masked -> exactly 0, pre-softmax)
__global__ void __launch_bounds__(256)
scores_kernel(const bf16_t* __restrict__ Q, const bf16_t* __restrict__ Km,
              const int* __restrict__ positions, float* __restrict__ scores) {
  constexpr int MT = 2, NT = 4;
  int w = blockIdx.x * 8 + (int)(threadIdx.x >> 5);
  constexpr int tilesPerB = (PP / (16 * MT)) * (SS / (16 * NT)); // 8*32 = 256
  int b = w / tilesPerB, rem = w % tilesPerB;
  int m0 = (rem / (SS / (16 * NT))) * 16 * MT;
  int n0 = (rem % (SS / (16 * NT))) * 16 * NT;
  const bf16_t* Qb = Q + (size_t)b * PP * UU;
  const bf16_t* Kb = Km + (size_t)b * SS * UU;
  v8f c[MT][NT] = {};
  gemm_core<MT, NT>(Qb, UU, Kb, UU, UU, m0, n0, c);
  int l = lane_id();
  int nn = l & 15, g = l >> 4;
#pragma unroll
  for (int i = 0; i < MT; ++i) {
#pragma unroll
    for (int r = 0; r < 8; ++r) {
      int m = m0 + 16 * i + r + 8 * g;
      int qmod = positions[b * PP + m] & (DIL - 1);
      float* orow = scores + ((size_t)b * PP + m) * SS;
#pragma unroll
      for (int j = 0; j < NT; ++j) {
        int t = n0 + 16 * j + nn;
        orow[t] = ((t & (DIL - 1)) == qmod) ? c[i][j][r] * SCALE : 0.0f;
      }
    }
  }
}

// row softmax over S, write bf16 probabilities
__global__ void __launch_bounds__(256)
softmax_rows(const float* __restrict__ scores, bf16_t* __restrict__ att) {
  int row = blockIdx.x;
  const float* r = scores + (size_t)row * SS;
  bf16_t* o = att + (size_t)row * SS;
  __shared__ float red[256];
  float mx = -3.0e38f;
  for (int t = threadIdx.x; t < SS; t += 256) mx = fmaxf(mx, r[t]);
  red[threadIdx.x] = mx; __syncthreads();
  for (int s = 128; s > 0; s >>= 1) {
    if ((int)threadIdx.x < s) red[threadIdx.x] = fmaxf(red[threadIdx.x], red[threadIdx.x + s]);
    __syncthreads();
  }
  mx = red[0]; __syncthreads();
  float sum = 0.0f;
  for (int t = threadIdx.x; t < SS; t += 256) sum += __expf(r[t] - mx);
  red[threadIdx.x] = sum; __syncthreads();
  for (int s = 128; s > 0; s >>= 1) {
    if ((int)threadIdx.x < s) red[threadIdx.x] += red[threadIdx.x + s];
    __syncthreads();
  }
  float inv = 1.0f / red[0];
  for (int t = threadIdx.x; t < SS; t += 256) o[t] = (bf16_t)(__expf(r[t] - mx) * inv);
}

// out[b][p][f] = att[b][p][:] @ x[b][:][f]   (B operand from xT [b][f][t], N-major)
__global__ void __launch_bounds__(256)
out_gemm(const bf16_t* __restrict__ Att, const bf16_t* __restrict__ XT,
         float* __restrict__ outp) {
  constexpr int MT = 2, NT = 4;
  int w = blockIdx.x * 8 + (int)(threadIdx.x >> 5);
  constexpr int tilesPerB = (PP / (16 * MT)) * (FF / (16 * NT)); // 8*8 = 64
  int b = w / tilesPerB, rem = w % tilesPerB;
  int m0 = (rem / (FF / (16 * NT))) * 16 * MT;
  int n0 = (rem % (FF / (16 * NT))) * 16 * NT;
  const bf16_t* Ab = Att + (size_t)b * PP * SS;
  const bf16_t* Xb = XT + (size_t)b * FF * SS;
  v8f c[MT][NT] = {};
  gemm_core<MT, NT>(Ab, SS, Xb, SS, SS, m0, n0, c);
  int l = lane_id();
  int g = l >> 4;
#pragma unroll
  for (int j = 0; j < NT; ++j) {
    int n = n0 + 16 * j + (l & 15);
#pragma unroll
    for (int i = 0; i < MT; ++i)
#pragma unroll
      for (int r = 0; r < 8; ++r) {
        int m = m0 + 16 * i + r + 8 * g;
        outp[((size_t)b * PP + m) * FF + n] = c[i][j][r];
      }
  }
}

extern "C" void kernel_launch(void* const* d_in, const int* in_sizes, int n_in,
                              void* d_out, int out_size, void* d_ws, size_t ws_size,
                              hipStream_t stream) {
  const float* x         = (const float*)d_in[0];
  const int*   positions = (const int*)d_in[1];
  const float* qw        = (const float*)d_in[2];
  const float* kw        = (const float*)d_in[3];
  const float* qb        = (const float*)d_in[4];
  const float* kb        = (const float*)d_in[5];
  float* outp = (float*)d_out;

  char* ws = (char*)d_ws;
  size_t off = 0;
  bf16_t* xbf  = (bf16_t*)(ws + off); off += (size_t)BB * SS * FF * 2;  // 16.8 MB
  bf16_t* xT   = (bf16_t*)(ws + off); off += (size_t)BB * FF * SS * 2;  // 16.8 MB
  bf16_t* kbf  = (bf16_t*)(ws + off); off += (size_t)BB * SS * UU * 2;  // 16.8 MB
  bf16_t* xq   = (bf16_t*)(ws + off); off += (size_t)BB * PP * FF * 2;  //  2.1 MB
  bf16_t* qp   = (bf16_t*)(ws + off); off += (size_t)BB * PP * UU * 2;  //  2.1 MB
  bf16_t* kwT  = (bf16_t*)(ws + off); off += (size_t)FF * UU * 2;       //  0.5 MB
  bf16_t* qwT  = (bf16_t*)(ws + off); off += (size_t)FF * UU * 2;       //  0.5 MB
  float*  scr  = (float*)(ws + off);  off += (size_t)BB * PP * SS * 4;  // 16.8 MB
  bf16_t* att  = (bf16_t*)(ws + off); off += (size_t)BB * PP * SS * 2;  //  8.4 MB

  // Prep: bf16 repacks / transposes / gather
  pack_x_kernel<<<4096, 256, 0, stream>>>(x, xbf, (size_t)BB * SS * FF);
  transpose_x_kernel<<<4096, 256, 0, stream>>>(x, xT);
  transpose_w_kernel<<<512, 256, 0, stream>>>(kw, kwT, FF, UU);
  transpose_w_kernel<<<512, 256, 0, stream>>>(qw, qwT, FF, UU);
  gather_xq_kernel<<<1024, 256, 0, stream>>>(x, positions, xq);

  // K projection: [B*S, U] = xbf @ kw + kb
  gemm_bias_bf16<<<(BB * SS / 32) * (UU / 64) / 8, 256, 0, stream>>>(xbf, kwT, kb, kbf, BB * SS);
  // Q projection at gathered rows: [B*P, U] = xq @ qw + qb
  gemm_bias_bf16<<<(BB * PP / 32) * (UU / 64) / 8, 256, 0, stream>>>(xq, qwT, qb, qp, BB * PP);
  // Masked/scaled scores [B,P,S]
  scores_kernel<<<BB * (PP / 32) * (SS / 64) / 8, 256, 0, stream>>>(qp, kbf, positions, scr);
  // Row softmax -> bf16 probabilities
  softmax_rows<<<BB * PP, 256, 0, stream>>>(scr, att);
  // out[b,p,:] = att @ x  (gather already applied)
  out_gemm<<<BB * (PP / 32) * (FF / 64) / 8, 256, 0, stream>>>(att, xT, outp);
}
